// BioTagger_8478265442485
// MI455X (gfx1250) — compile-verified, matrix operand order checked by reading, not run on hardware
//
#include <hip/hip_runtime.h>
#include <hip/hip_bf16.h>
#include <math.h>

// ---------------- problem constants ----------------
#define Bn    128
#define Sn    256
#define EMBn  300
#define Hn    512
#define Tn    9
#define KPAD  320                 // EMB padded to multiple of 32
#define NROWS (Bn * Sn)           // 32768
#define G4H   (4 * Hn)            // 2048

// ---------------- CDNA5 WMMA plumbing ----------------
typedef __bf16 bf16_t;
typedef __attribute__((ext_vector_type(16))) __bf16 v16bf;
typedef __attribute__((ext_vector_type(8)))  float  v8f;

union AccU  { v8f  v; float  e[8];  };
union FragU { v16bf v; bf16_t e[16]; };

__device__ __forceinline__ float sigmoidf_(float x) { return 1.0f / (1.0f + __expf(-x)); }
__device__ __forceinline__ bf16_t f2bf(float x) { return (bf16_t)x; }

// 16x32 bf16 fragment from a row-major matrix (rows = M for A-frag, rows = N for
// B-frag since B = W^T with W row-major). CDNA5 ISA 16-bit layout:
// lanes 0-15 : elems 0..7 = K 0..7,  elems 8..15 = K 16..23
// lanes 16-31: elems 0..7 = K 8..15, elems 8..15 = K 24..31
__device__ __forceinline__ v16bf load_frag(const bf16_t* __restrict__ base, int lda, int lane) {
  const int half = (lane >> 4) & 1;
  const int r    = lane & 15;
  const bf16_t* p = base + (size_t)r * lda + half * 8;
  FragU u;
#pragma unroll
  for (int i = 0; i < 8; ++i) { u.e[i] = p[i]; u.e[8 + i] = p[16 + i]; }
  return u.v;
}

__device__ __forceinline__ v8f wmma_bf16(v16bf a, v16bf b, v8f c) {
  // 8 args: (neg_a, A, neg_b, B, c_mod, C, reuse_a, reuse_b)
  return __builtin_amdgcn_wmma_f32_16x16x32_bf16(false, a, false, b, (short)0, c, false, false);
}

__device__ __forceinline__ v8f zero8() {
  v8f z = {0.f, 0.f, 0.f, 0.f, 0.f, 0.f, 0.f, 0.f};
  return z;
}

// ---------------- K0: weight conversion / padding to bf16 ----------------
__global__ void convert_weights_kernel(const float* __restrict__ wihf, const float* __restrict__ wihb,
                                       const float* __restrict__ whhf, const float* __restrict__ whhb,
                                       const float* __restrict__ fcw,
                                       bf16_t* __restrict__ Wih,   // [2][2048][320]
                                       bf16_t* __restrict__ Whh,   // [2][2048][512]
                                       bf16_t* __restrict__ Fcw) { // [16][1024]
  const int tid = blockIdx.x * blockDim.x + threadIdx.x;
  const int stride = gridDim.x * blockDim.x;
  const int tot1 = 2 * G4H * KPAD;
  for (int i = tid; i < tot1; i += stride) {
    int dir = i / (G4H * KPAD);
    int rem = i - dir * G4H * KPAD;
    int r = rem / KPAD, k = rem % KPAD;
    const float* src = dir ? wihb : wihf;
    Wih[i] = f2bf(k < EMBn ? src[(size_t)r * EMBn + k] : 0.f);
  }
  const int tot2 = 2 * G4H * Hn;
  for (int i = tid; i < tot2; i += stride) {
    int dir = i / (G4H * Hn);
    int rem = i - dir * G4H * Hn;
    Whh[i] = f2bf((dir ? whhb : whhf)[rem]);
  }
  const int tot3 = 16 * (2 * Hn);
  for (int i = tid; i < tot3; i += stride) {
    int r = i / (2 * Hn), k = i % (2 * Hn);
    Fcw[i] = f2bf(r < Tn ? fcw[(size_t)r * 2 * Hn + k] : 0.f);
  }
}

// ---------------- K1: embedding gather -> bf16 (K-padded) ----------------
__global__ void embed_kernel(const int* __restrict__ inputs, const float* __restrict__ emb,
                             bf16_t* __restrict__ X) { // [32768][320]
  const int tid = blockIdx.x * blockDim.x + threadIdx.x;
  const int stride = gridDim.x * blockDim.x;
  const int tot = NROWS * KPAD;
  for (int i = tid; i < tot; i += stride) {
    int row = i / KPAD, k = i % KPAD;
    float v = 0.f;
    if (k < EMBn) v = emb[(size_t)inputs[row] * EMBn + k];
    X[i] = f2bf(v);
  }
}

// ---------------- Kinit: zero h (both parities) and c ----------------
__global__ void init_state_kernel(bf16_t* __restrict__ hbf, float* __restrict__ c) {
  const int tid = blockIdx.x * blockDim.x + threadIdx.x;
  const int stride = gridDim.x * blockDim.x;
  const int toth = 2 * 2 * Bn * Hn;        // [parity][dir][B][H]
  for (int i = tid; i < toth; i += stride) hbf[i] = f2bf(0.f);
  const int totc = 2 * Bn * Hn;
  for (int i = tid; i < totc; i += stride) c[i] = 0.f;
}

// ---------------- K2: xg = X * W_ih^T + b  (both dirs) ----------------
// One wave computes a 16(M) x 64(N) tile: 4 accumulators, A-frag reused 4x.
__global__ void gates_gemm_kernel(const bf16_t* __restrict__ X,    // [32768][320]
                                  const bf16_t* __restrict__ Wih,  // [2][2048][320]
                                  const float* __restrict__ bias_f,
                                  const float* __restrict__ bias_b,
                                  float* __restrict__ xg) {        // [2][32768][2048]
  const int dir  = blockIdx.y;
  const int lane = threadIdx.x & 31;
  const int wave = threadIdx.x >> 5;
  const int gid  = blockIdx.x * 8 + wave;     // 0..65535
  const int mt   = gid >> 5;                  // 0..2047
  const int nt   = gid & 31;                  // 0..31
  const int m0   = mt * 16, n0 = nt * 64;
  const bf16_t* w = Wih + (size_t)dir * G4H * KPAD;

  v8f acc0 = zero8(), acc1 = zero8(), acc2 = zero8(), acc3 = zero8();
  for (int kk = 0; kk < KPAD; kk += 32) {
    v16bf a = load_frag(X + (size_t)m0 * KPAD + kk, KPAD, lane);
    v16bf b0 = load_frag(w + (size_t)(n0 +  0) * KPAD + kk, KPAD, lane);
    acc0 = wmma_bf16(a, b0, acc0);
    v16bf b1 = load_frag(w + (size_t)(n0 + 16) * KPAD + kk, KPAD, lane);
    acc1 = wmma_bf16(a, b1, acc1);
    v16bf b2 = load_frag(w + (size_t)(n0 + 32) * KPAD + kk, KPAD, lane);
    acc2 = wmma_bf16(a, b2, acc2);
    v16bf b3 = load_frag(w + (size_t)(n0 + 48) * KPAD + kk, KPAD, lane);
    acc3 = wmma_bf16(a, b3, acc3);
  }
  const float* bias = dir ? bias_b : bias_f;
  AccU u[4]; u[0].v = acc0; u[1].v = acc1; u[2].v = acc2; u[3].v = acc3;
  float* out = xg + (size_t)dir * NROWS * G4H;
#pragma unroll
  for (int sub = 0; sub < 4; ++sub) {
    const int n = n0 + sub * 16 + (lane & 15);
#pragma unroll
    for (int v = 0; v < 8; ++v) {
      const int m = m0 + ((lane >> 4) << 3) + v;
      out[(size_t)m * G4H + n] = u[sub].e[v] + bias[n];
    }
  }
}

// ---------------- K3: one LSTM timestep, both directions, gates fused ----------------
// One wave owns a 16(batch) x 16(hidden-j) tile and runs 4 WMMA chains (i,f,g,o),
// so the nonlinear c/h update is done fully in-register.
__global__ void lstm_step_kernel(const float* __restrict__ xg,    // [2][32768][2048]
                                 const bf16_t* __restrict__ whh,  // [2][2048][512]
                                 const bf16_t* __restrict__ hin,  // [2][128][512] (parity t)
                                 bf16_t* __restrict__ hout,       // [2][128][512] (parity t+1)
                                 float* __restrict__ c,           // [2][128][512]
                                 bf16_t* __restrict__ feats,      // [32768][1024]
                                 int t) {
  const int dir  = blockIdx.y;
  const int lane = threadIdx.x & 31;
  const int wave = threadIdx.x >> 5;
  const int tile = blockIdx.x * 8 + wave;     // 0..255
  const int mt   = tile >> 5;                 // 0..7   (batch tiles)
  const int jt   = tile & 31;                 // 0..31  (hidden tiles)
  const int m0   = mt * 16, j0 = jt * 16;

  const bf16_t* hprev = hin + (size_t)dir * Bn * Hn;
  const bf16_t* w     = whh + (size_t)dir * G4H * Hn;

  v8f acc0 = zero8(), acc1 = zero8(), acc2 = zero8(), acc3 = zero8();
  for (int kk = 0; kk < Hn; kk += 32) {
    v16bf a = load_frag(hprev + (size_t)m0 * Hn + kk, Hn, lane);
    v16bf b0 = load_frag(w + (size_t)(0 * Hn + j0) * Hn + kk, Hn, lane);
    acc0 = wmma_bf16(a, b0, acc0);
    v16bf b1 = load_frag(w + (size_t)(1 * Hn + j0) * Hn + kk, Hn, lane);
    acc1 = wmma_bf16(a, b1, acc1);
    v16bf b2 = load_frag(w + (size_t)(2 * Hn + j0) * Hn + kk, Hn, lane);
    acc2 = wmma_bf16(a, b2, acc2);
    v16bf b3 = load_frag(w + (size_t)(3 * Hn + j0) * Hn + kk, Hn, lane);
    acc3 = wmma_bf16(a, b3, acc3);
  }

  const int s = dir ? (Sn - 1 - t) : t;       // backward dir scans reversed
  AccU ui, uf, ug, uo;
  ui.v = acc0; uf.v = acc1; ug.v = acc2; uo.v = acc3;
  const float* xgd = xg + (size_t)dir * NROWS * G4H;
  const int j = j0 + (lane & 15);
#pragma unroll
  for (int v = 0; v < 8; ++v) {
    const int m = m0 + ((lane >> 4) << 3) + v;
    const size_t row   = (size_t)m * Sn + s;
    const size_t gbase = row * G4H;
    const float gi = ui.e[v] + xgd[gbase + 0 * Hn + j];
    const float gf = uf.e[v] + xgd[gbase + 1 * Hn + j];
    const float gg = ug.e[v] + xgd[gbase + 2 * Hn + j];
    const float go = uo.e[v] + xgd[gbase + 3 * Hn + j];
    const size_t hidx = (size_t)dir * Bn * Hn + (size_t)m * Hn + j;
    const float cn = sigmoidf_(gf) * c[hidx] + sigmoidf_(gi) * tanhf(gg);
    const float h  = sigmoidf_(go) * tanhf(cn);
    c[hidx] = cn;
    hout[hidx] = f2bf(h);
    feats[row * (2 * Hn) + (size_t)dir * Hn + j] = f2bf(h);
  }
}

// ---------------- K4: emissions = feats * fc_w^T (T padded to 16) ----------------
__global__ void emissions_kernel(const bf16_t* __restrict__ feats, // [32768][1024]
                                 const bf16_t* __restrict__ fcw,   // [16][1024]
                                 float* __restrict__ emis) {       // [32768][16]
  const int lane = threadIdx.x & 31;
  const int wave = threadIdx.x >> 5;
  const int tile = blockIdx.x * 8 + wave;     // 0..2047
  const int m0   = tile * 16;
  v8f acc = zero8();
  for (int kk = 0; kk < 2 * Hn; kk += 32) {
    v16bf a = load_frag(feats + (size_t)m0 * (2 * Hn) + kk, 2 * Hn, lane);
    v16bf b = load_frag(fcw + kk, 2 * Hn, lane);
    acc = wmma_bf16(a, b, acc);
  }
  AccU u; u.v = acc;
  const int n = lane & 15;
#pragma unroll
  for (int v = 0; v < 8; ++v) {
    const int m = m0 + ((lane >> 4) << 3) + v;
    emis[(size_t)m * 16 + n] = u.e[v];
  }
}

// ---------------- K5: CRF log-likelihood (numerator + forward algorithm) ----------------
__global__ void crf_kernel(const float* __restrict__ emis,   // [B][S][16]
                           const int* __restrict__ tags,     // [B][S]
                           const unsigned char* __restrict__ masks, // [B][S] (bool)
                           const float* __restrict__ trans,  // [9][9]
                           const float* __restrict__ strans, // [9]
                           const float* __restrict__ etrans, // [9]
                           float* __restrict__ out) {
  __shared__ float red[Bn];
  const int b = threadIdx.x;                  // one sequence per thread, 128 threads
  const float* em = emis + (size_t)b * Sn * 16;
  const int* tg = tags + (size_t)b * Sn;
  const unsigned char* mk = masks + (size_t)b * Sn;

  // numerator
  float num = strans[tg[0]] + em[tg[0]];
  int total = mk[0] ? 1 : 0;
  for (int s = 1; s < Sn; ++s) {
    const float m = mk[s] ? 1.f : 0.f;
    total += mk[s] ? 1 : 0;
    num += (trans[tg[s - 1] * Tn + tg[s]] + em[(size_t)s * 16 + tg[s]]) * m;
  }
  const int lastidx = total > 0 ? total - 1 : 0;
  num += etrans[tg[lastidx]];

  // partition function (forward algorithm)
  float sc[Tn];
#pragma unroll
  for (int j = 0; j < Tn; ++j) sc[j] = strans[j] + em[j];
  for (int s = 1; s < Sn; ++s) {
    if (mk[s]) {
      float ns[Tn];
#pragma unroll
      for (int j = 0; j < Tn; ++j) {
        float mx = -3.4e38f;
#pragma unroll
        for (int i = 0; i < Tn; ++i) { float v = sc[i] + trans[i * Tn + j]; mx = fmaxf(mx, v); }
        float sum = 0.f;
#pragma unroll
        for (int i = 0; i < Tn; ++i) sum += __expf(sc[i] + trans[i * Tn + j] - mx);
        ns[j] = mx + __logf(sum) + em[(size_t)s * 16 + j];
      }
#pragma unroll
      for (int j = 0; j < Tn; ++j) sc[j] = ns[j];
    }
  }
  float mx = -3.4e38f;
#pragma unroll
  for (int j = 0; j < Tn; ++j) mx = fmaxf(mx, sc[j] + etrans[j]);
  float sum = 0.f;
#pragma unroll
  for (int j = 0; j < Tn; ++j) sum += __expf(sc[j] + etrans[j] - mx);
  const float logZ = mx + __logf(sum);

  red[b] = num - logZ;
  __syncthreads();
  for (int st = Bn / 2; st > 0; st >>= 1) {
    if (b < st) red[b] += red[b + st];
    __syncthreads();
  }
  if (b == 0) out[0] = red[0];
}

// ---------------- workspace layout ----------------
static constexpr size_t SZ_XBF   = (size_t)NROWS * KPAD * 2;        //  20.97 MB  X bf16
static constexpr size_t SZ_WIH   = (size_t)2 * G4H * KPAD * 2;      //   2.62 MB
static constexpr size_t SZ_WHH   = (size_t)2 * G4H * Hn * 2;        //   4.19 MB
static constexpr size_t SZ_FCW   = (size_t)16 * 2 * Hn * 2;         //   32 KB
static constexpr size_t SZ_XG    = (size_t)2 * NROWS * G4H * 4;     // 536.87 MB
static constexpr size_t SZ_HBF   = (size_t)2 * 2 * Bn * Hn * 2;     // 512 KB (double-buffered)
static constexpr size_t SZ_C     = (size_t)2 * Bn * Hn * 4;         // 512 KB
static constexpr size_t SZ_FEATS = (size_t)NROWS * 2 * Hn * 2;      //  67.1 MB
// emissions after feats

extern "C" void kernel_launch(void* const* d_in, const int* in_sizes, int n_in,
                              void* d_out, int out_size, void* d_ws, size_t ws_size,
                              hipStream_t stream) {
  const int*   inputs = (const int*)d_in[0];
  const int*   tags   = (const int*)d_in[1];
  const unsigned char* masks = (const unsigned char*)d_in[2];
  const float* emb    = (const float*)d_in[3];
  const float* w_ih_f = (const float*)d_in[4];
  const float* w_hh_f = (const float*)d_in[5];
  const float* b_f    = (const float*)d_in[6];
  const float* w_ih_b = (const float*)d_in[7];
  const float* w_hh_b = (const float*)d_in[8];
  const float* b_b    = (const float*)d_in[9];
  const float* fc_w   = (const float*)d_in[10];
  const float* trans  = (const float*)d_in[11];
  const float* strans = (const float*)d_in[12];
  const float* etrans = (const float*)d_in[13];
  float* out = (float*)d_out;

  char* ws = (char*)d_ws;
  bf16_t* Xbf   = (bf16_t*)(ws);                          ws += SZ_XBF;
  bf16_t* Wih   = (bf16_t*)(ws);                          ws += SZ_WIH;
  bf16_t* Whh   = (bf16_t*)(ws);                          ws += SZ_WHH;
  bf16_t* Fcw   = (bf16_t*)(ws);                          ws += SZ_FCW;
  float*  xg    = (float*)(ws);                           ws += SZ_XG;
  bf16_t* hbf   = (bf16_t*)(ws);                          ws += SZ_HBF;
  float*  cstate= (float*)(ws);                           ws += SZ_C;
  bf16_t* feats = (bf16_t*)(ws);                          ws += SZ_FEATS;
  float*  emis  = (float*)(ws);

  // K0/K1/Kinit: conversions, embedding gather, state reset (every call, deterministic)
  convert_weights_kernel<<<4096, 256, 0, stream>>>(w_ih_f, w_ih_b, w_hh_f, w_hh_b, fc_w,
                                                   Wih, Whh, Fcw);
  embed_kernel<<<8192, 256, 0, stream>>>(inputs, emb, Xbf);
  init_state_kernel<<<512, 256, 0, stream>>>(hbf, cstate);

  // K2: input projections for both directions (WMMA bf16)
  gates_gemm_kernel<<<dim3(8192, 2), 256, 0, stream>>>(Xbf, Wih, b_f, b_b, xg);

  // K3: 256 sequential LSTM steps (fwd + bwd fused per launch), h double-buffered
  const size_t HPAR = (size_t)2 * Bn * Hn;   // elements per parity
  for (int t = 0; t < Sn; ++t) {
    const bf16_t* hin  = hbf + (size_t)(t & 1) * HPAR;
    bf16_t*       hout = hbf + (size_t)((t + 1) & 1) * HPAR;
    lstm_step_kernel<<<dim3(32, 2), 256, 0, stream>>>(xg, Whh, hin, hout, cstate, feats, t);
  }

  // K4: emissions GEMM (WMMA, T padded to 16)
  emissions_kernel<<<256, 256, 0, stream>>>(feats, Fcw, emis);

  // K5: CRF log-likelihood -> scalar
  crf_kernel<<<1, Bn, 0, stream>>>(emis, tags, masks, trans, strans, etrans, out);
}
